// GraphAttentionLayer_88510686036621
// MI455X (gfx1250) — matrix operand order repaired and checked by reading, not compiled, not run on hardware
//
#include <hip/hip_runtime.h>

#define BB 8
#define NN 2048
#define DD 256
#define HH 8
#define HD 32
#define NT (NN/16)   // 128 query tiles per batch

typedef __attribute__((ext_vector_type(16))) __bf16 v16bf;
typedef __attribute__((ext_vector_type(8)))  float  v8f;

struct U128x2 { uint4 lo, hi; };

static __device__ __forceinline__ v16bf pack_bf16(uint4 lo, uint4 hi) {
  U128x2 t{lo, hi};
  return __builtin_bit_cast(v16bf, t);
}

static __device__ __forceinline__ unsigned short f2bf(float f) {
  unsigned u = __float_as_uint(f);
  u += 0x7fffu + ((u >> 16) & 1u);   // round-to-nearest-even
  return (unsigned short)(u >> 16);
}

static __device__ __forceinline__ v8f vzero8() {
  v8f z = {0.f,0.f,0.f,0.f,0.f,0.f,0.f,0.f};
  return z;
}

static __device__ __forceinline__ v8f wmma_bf16(v16bf a, v16bf b, v8f c) {
  // D = A(16x32 bf16) x B(32x16 bf16) + C(16x16 f32)
  return __builtin_amdgcn_wmma_f32_16x16x32_bf16(false, a, false, b, (short)0, c,
                                                 false, false);
}

// ---------------------------------------------------------------- prep kernels
__global__ void k_cvt_x(const float* __restrict__ x, unsigned short* __restrict__ xb, int n) {
  for (int i = blockIdx.x * blockDim.x + threadIdx.x; i < n; i += gridDim.x * blockDim.x)
    xb[i] = f2bf(x[i]);
}

// Pre-swizzle the 4 weight matrices into WMMA B-operand K-panels:
// panel[(k/32)*D*32 + col*32 + (k%32)] = bf16(W[k*D + col])
// -> each lane's 16 B-values (K 0..15 or 16..31 of one column) are contiguous.
__global__ void k_panels(const float* __restrict__ wq, const float* __restrict__ wk,
                         const float* __restrict__ wv, const float* __restrict__ wo,
                         unsigned short* __restrict__ pq, unsigned short* __restrict__ pk,
                         unsigned short* __restrict__ pv, unsigned short* __restrict__ po) {
  int i = blockIdx.x * blockDim.x + threadIdx.x;   // 0 .. 4*D*D
  int m = i >> 16;
  int e = i & 65535;
  int k = e >> 8, c = e & 255;
  const float* src = (m == 0) ? wq : (m == 1) ? wk : (m == 2) ? wv : wo;
  unsigned short* dst = (m == 0) ? pq : (m == 1) ? pk : (m == 2) ? pv : po;
  dst[(size_t)(k >> 5) * (DD*32) + (size_t)c * 32 + (k & 31)] = f2bf(src[(size_t)k*DD + c]);
}

// ---------------------------------------------------------------- QKV GEMM
// One block = one (b, 16-row tile of N). 8 waves share the x tile via LDS and
// cover 3 matrices x 16 column tiles (6 output tiles per wave, 8 WMMAs each).
__global__ __launch_bounds__(256) void k_qkv(
    const unsigned short* __restrict__ xb,
    const unsigned short* __restrict__ pq, const unsigned short* __restrict__ pk,
    const unsigned short* __restrict__ pv,
    const float* __restrict__ bq, const float* __restrict__ bk, const float* __restrict__ bv,
    unsigned short* __restrict__ qh, unsigned short* __restrict__ kh,
    unsigned short* __restrict__ vt)
{
  __shared__ __align__(16) unsigned short xs[16*DD];   // 8 KB x-tile (bf16)
  int blk = blockIdx.x;
  int b = blk >> 7, n0 = (blk & 127) * 16;
  int tid = threadIdx.x, w = tid >> 5, lane = tid & 31;
  int hf = lane >> 4, l16 = lane & 15;
  int rowoff = hf ? 8 : 0;         // C-layout: lanes 16..31 hold rows M+8
  int koff   = hf ? 16 : 0;        // A-layout: lanes 16..31 hold K 16..31
  int bkoff  = hf ? 16 : 0;        // B-layout: lanes 16..31 hold K 16..31

  { // cooperative 8 KB contiguous copy
    const uint4* s = (const uint4*)(xb + (size_t)(b*NN + n0) * DD);
    uint4* d = (uint4*)xs;
    d[tid] = s[tid]; d[tid + 256] = s[tid + 256];
  }
  __syncthreads();

  for (int it = 0; it < 6; ++it) {
    int p = w + 8*it, mat = p >> 4, ct = p & 15, cbase = ct*16;
    const unsigned short* wp = (mat == 0) ? pq : (mat == 1) ? pk : pv;
    const float* bias = (mat == 0) ? bq : (mat == 1) ? bk : bv;
    v8f c = vzero8();
    #pragma unroll
    for (int kt = 0; kt < 8; ++kt) {
      const unsigned short* arow = xs + l16*DD + kt*32 + koff;
      v16bf A  = pack_bf16(*(const uint4*)arow, *(const uint4*)(arow + 8));
      const unsigned short* brow = wp + ((size_t)kt*DD + (cbase + l16)) * 32 + bkoff;
      v16bf Bm = pack_bf16(*(const uint4*)brow, *(const uint4*)(brow + 8));
      c = wmma_bf16(A, Bm, c);
    }
    float bb = bias[cbase + l16];
    int colg = cbase + l16;          // global output column
    int h = colg >> 5, d = colg & 31;
    #pragma unroll
    for (int r = 0; r < 8; ++r) {
      int n = n0 + r + rowoff;
      unsigned short v16 = f2bf(c[r] + bb);
      if (mat == 0)      qh[((size_t)(b*HH + h)*NN + n)*HD + d] = v16;   // [B,H,N,HD]
      else if (mat == 1) kh[((size_t)(b*HH + h)*NN + n)*HD + d] = v16;   // [B,H,N,HD]
      else               vt[((size_t)(b*HH + h)*HD + d)*NN + n] = v16;   // [B,H,HD,N]
    }
  }
}

// ---------------------------------------------------------------- attention
// One block = (b, 16-query tile); wave w = head w. Adjacency tile is loaded to
// LDS once per 32-key step and shared by all 8 heads (adjacency read exactly
// once from HBM). Flash-style online softmax, all GEMMs via WMMA.
__global__ __launch_bounds__(256) void k_attn(
    const unsigned short* __restrict__ qh, const unsigned short* __restrict__ kh,
    const unsigned short* __restrict__ vt, const int* __restrict__ adj,
    unsigned short* __restrict__ att)
{
  __shared__ int adj_s[16*32];                              // 2 KB mask tile
  __shared__ __align__(16) unsigned short p_s[8][16*32];    // 1 KB/wave P staging
  const float SCALE = 0.17677669529663687f;                 // 1/sqrt(HD)
  const float NEG = -__builtin_inff();

  int blk = blockIdx.x;
  int b = blk >> 7, q0 = (blk & 127) * 16;
  int tid = threadIdx.x, w = tid >> 5, lane = tid & 31;
  int hf = lane >> 4, l16 = lane & 15;
  int rowoff = hf ? 8 : 0, koff = hf ? 16 : 0, bkoff = hf ? 16 : 0;
  int h = w;

  // Q tile in A-layout: lane row = l16, K span per half
  const unsigned short* qrow = qh + ((size_t)(b*HH + h)*NN + q0 + l16) * HD + koff;
  v16bf QA = pack_bf16(*(const uint4*)qrow, *(const uint4*)(qrow + 8));

  const unsigned short* kbase = kh + (size_t)(b*HH + h) * NN * HD;
  const unsigned short* vbase = vt + (size_t)(b*HH + h) * HD * NN;
  unsigned short* pw = &p_s[w][0];
  const int* adjrow0 = adj + ((size_t)b*NN + q0) * NN;

  float m[8], l[8];
  #pragma unroll
  for (int r = 0; r < 8; ++r) { m[r] = NEG; l[r] = 0.f; }
  v8f acc0 = vzero8(), acc1 = vzero8();

  for (int kt = 0; kt < 64; ++kt) {
    int t = kt * 32;
    { // cooperative adjacency tile: adj_s[r*32+c] = adj[b][q0+r][t+c]
      int e0 = tid, e1 = tid + 256;
      adj_s[e0] = adjrow0[(size_t)(e0 >> 5) * NN + t + (e0 & 31)];
      adj_s[e1] = adjrow0[(size_t)(e1 >> 5) * NN + t + (e1 & 31)];
      if (kt + 1 < 64)
        __builtin_prefetch(&adjrow0[(size_t)(e0 >> 5) * NN + t + 32 + (e0 & 31)], 0, 1);
    }
    __syncthreads();

    // scores: Q(16x32) x K^T(32x16) per 16-key half; K rows are B-columns
    const unsigned short* k0p = kbase + (size_t)(t + l16) * HD + bkoff;
    const unsigned short* k1p = kbase + (size_t)(t + 16 + l16) * HD + bkoff;
    v16bf KB0 = pack_bf16(*(const uint4*)k0p, *(const uint4*)(k0p + 8));
    v16bf KB1 = pack_bf16(*(const uint4*)k1p, *(const uint4*)(k1p + 8));
    v8f s0 = wmma_bf16(QA, KB0, vzero8());
    v8f s1 = wmma_bf16(QA, KB1, vzero8());

    // online softmax over the 32 new keys (row lives in a 16-lane half)
    #pragma unroll
    for (int r = 0; r < 8; ++r) {
      int arow = r + rowoff;
      float sc0 = s0[r] * SCALE;
      float sc1 = s1[r] * SCALE;
      if (adj_s[arow*32 + l16] == 0)      sc0 = NEG;
      if (adj_s[arow*32 + 16 + l16] == 0) sc1 = NEG;
      float rm = fmaxf(sc0, sc1);
      rm = fmaxf(rm, __shfl_xor(rm, 1, 32));
      rm = fmaxf(rm, __shfl_xor(rm, 2, 32));
      rm = fmaxf(rm, __shfl_xor(rm, 4, 32));
      rm = fmaxf(rm, __shfl_xor(rm, 8, 32));
      float mn = fmaxf(m[r], rm);
      float al = (m[r] == NEG) ? 0.f : __expf(m[r] - mn);
      float p0 = (sc0 == NEG) ? 0.f : __expf(sc0 - mn);
      float p1 = (sc1 == NEG) ? 0.f : __expf(sc1 - mn);
      float rs = p0 + p1;
      rs += __shfl_xor(rs, 1, 32);
      rs += __shfl_xor(rs, 2, 32);
      rs += __shfl_xor(rs, 4, 32);
      rs += __shfl_xor(rs, 8, 32);
      l[r] = l[r] * al + rs;
      m[r] = mn;
      acc0[r] *= al;
      acc1[r] *= al;
      pw[arow*32 + l16]      = f2bf(p0);   // C-layout -> LDS (row-major P)
      pw[arow*32 + 16 + l16] = f2bf(p1);
    }

    // P back in A-layout (same-wave DS ordering guarantees visibility)
    const unsigned short* prow = pw + l16*32 + koff;
    v16bf PA = pack_bf16(*(const uint4*)prow, *(const uint4*)(prow + 8));

    // V tiles in B-layout: columns are dims, K are the 32 keys
    const unsigned short* v0p = vbase + (size_t)l16 * NN + t + bkoff;
    const unsigned short* v1p = vbase + (size_t)(16 + l16) * NN + t + bkoff;
    v16bf VB0 = pack_bf16(*(const uint4*)v0p, *(const uint4*)(v0p + 8));
    v16bf VB1 = pack_bf16(*(const uint4*)v1p, *(const uint4*)(v1p + 8));
    acc0 = wmma_bf16(PA, VB0, acc0);
    acc1 = wmma_bf16(PA, VB1, acc1);

    __syncthreads();   // protect adj_s before next step's stores
  }

  #pragma unroll
  for (int r = 0; r < 8; ++r) {
    float inv = 1.0f / l[r];
    int n = q0 + r + rowoff;
    att[((size_t)b*NN + n)*DD + h*32 + l16]      = f2bf(acc0[r] * inv);
    att[((size_t)b*NN + n)*DD + h*32 + 16 + l16] = f2bf(acc1[r] * inv);
  }
}

// ---------------------------------------------------------------- out-proj + residual + LN
__global__ __launch_bounds__(256) void k_out(
    const unsigned short* __restrict__ att, const unsigned short* __restrict__ po,
    const float* __restrict__ bo, const float* __restrict__ x,
    const float* __restrict__ gamma, const float* __restrict__ beta,
    float* __restrict__ out)
{
  __shared__ __align__(16) unsigned short as_[16*DD];  // 8 KB attended tile
  __shared__ float ys[16*DD];                          // 16 KB f32 GEMM result
  int blk = blockIdx.x;
  int b = blk >> 7, n0 = (blk & 127) * 16;
  int tid = threadIdx.x, w = tid >> 5, lane = tid & 31;
  int hf = lane >> 4, l16 = lane & 15;
  int rowoff = hf ? 8 : 0, koff = hf ? 16 : 0, bkoff = hf ? 16 : 0;

  { const uint4* s = (const uint4*)(att + (size_t)(b*NN + n0) * DD);
    uint4* d = (uint4*)as_;
    d[tid] = s[tid]; d[tid + 256] = s[tid + 256]; }
  __syncthreads();

  #pragma unroll
  for (int ph = 0; ph < 2; ++ph) {     // 16 col tiles over 8 waves
    int ct = w*2 + ph, cbase = ct*16;
    v8f c = vzero8();
    #pragma unroll
    for (int kt = 0; kt < 8; ++kt) {
      const unsigned short* arow = as_ + l16*DD + kt*32 + koff;
      v16bf A  = pack_bf16(*(const uint4*)arow, *(const uint4*)(arow + 8));
      const unsigned short* brow = po + ((size_t)kt*DD + (cbase + l16)) * 32 + bkoff;
      v16bf Bm = pack_bf16(*(const uint4*)brow, *(const uint4*)(brow + 8));
      c = wmma_bf16(A, Bm, c);
    }
    float bb = bo[cbase + l16];
    #pragma unroll
    for (int r = 0; r < 8; ++r)
      ys[(r + rowoff)*DD + cbase + l16] = c[r] + bb;
  }
  __syncthreads();

  // residual + LayerNorm: 16 lanes per row, 16 elems per lane
  int row = tid >> 4, j = tid & 15;
  const float* xrow = x + (size_t)(b*NN + n0 + row) * DD;
  float vals[16];
  float sum = 0.f;
  #pragma unroll
  for (int i = 0; i < 16; ++i) {
    float v = ys[row*DD + j*16 + i] + xrow[j*16 + i];
    vals[i] = v; sum += v;
  }
  sum += __shfl_xor(sum, 1, 32);
  sum += __shfl_xor(sum, 2, 32);
  sum += __shfl_xor(sum, 4, 32);
  sum += __shfl_xor(sum, 8, 32);
  float mu = sum * (1.0f/DD);
  float ss = 0.f;
  #pragma unroll
  for (int i = 0; i < 16; ++i) { float dd = vals[i] - mu; ss += dd*dd; }
  ss += __shfl_xor(ss, 1, 32);
  ss += __shfl_xor(ss, 2, 32);
  ss += __shfl_xor(ss, 4, 32);
  ss += __shfl_xor(ss, 8, 32);
  float rstd = rsqrtf(ss * (1.0f/DD) + 1e-5f);
  float* orow = out + (size_t)(b*NN + n0 + row) * DD;
  #pragma unroll
  for (int i = 0; i < 16; ++i) {
    int cc = j*16 + i;
    orow[cc] = (vals[i] - mu) * rstd * gamma[cc] + beta[cc];
  }
}

// ---------------------------------------------------------------- launch
extern "C" void kernel_launch(void* const* d_in, const int* in_sizes, int n_in,
                              void* d_out, int out_size, void* d_ws, size_t ws_size,
                              hipStream_t stream) {
  const float* x     = (const float*)d_in[0];
  const int*   adj   = (const int*)d_in[1];
  const float* Wq    = (const float*)d_in[2];
  const float* bq    = (const float*)d_in[3];
  const float* Wk    = (const float*)d_in[4];
  const float* bk    = (const float*)d_in[5];
  const float* Wv    = (const float*)d_in[6];
  const float* bv    = (const float*)d_in[7];
  const float* Wo    = (const float*)d_in[8];
  const float* bo    = (const float*)d_in[9];
  const float* gamma = (const float*)d_in[10];
  const float* beta  = (const float*)d_in[11];
  float* out = (float*)d_out;

  // workspace carve-up (bf16): x, Q[B,H,N,HD], K[B,H,N,HD], V^T[B,H,HD,N],
  // attended[B,N,D], 4 weight K-panels. Total ~41 MB.
  const size_t XE = (size_t)BB * NN * DD;          // 4,194,304 elems
  unsigned short* xb   = (unsigned short*)d_ws;
  unsigned short* qh   = xb   + XE;
  unsigned short* kh   = qh   + XE;
  unsigned short* vt   = kh   + XE;
  unsigned short* attb = vt   + XE;
  unsigned short* pq   = attb + XE;
  unsigned short* pk   = pq + (size_t)DD*DD;
  unsigned short* pv   = pk + (size_t)DD*DD;
  unsigned short* po   = pv + (size_t)DD*DD;

  k_cvt_x <<<2048, 256, 0, stream>>>(x, xb, (int)XE);
  k_panels<<<(4*DD*DD)/256, 256, 0, stream>>>(Wq, Wk, Wv, Wo, pq, pk, pv, po);
  k_qkv   <<<BB*NT, 256, 0, stream>>>(xb, pq, pk, pv, bq, bk, bv, qh, kh, vt);
  k_attn  <<<BB*NT, 256, 0, stream>>>(qh, kh, vt, adj, attb);
  k_out   <<<BB*NT, 256, 0, stream>>>(attb, po, bo, x, gamma, beta, out);
}
